// HandwritingSynthesis_87419764343370
// MI455X (gfx1250) — matrix operand order repaired and checked by reading, not compiled, not run on hardware
//
#include <hip/hip_runtime.h>
#include <hip/hip_bf16.h>
#include <math.h>

typedef __attribute__((ext_vector_type(16))) __bf16 v16bf;
typedef __attribute__((ext_vector_type(8)))  float  v8f;
typedef __attribute__((ext_vector_type(4)))  unsigned int u32x4;
typedef __attribute__((ext_vector_type(8)))  int          i32x8;
typedef __attribute__((ext_vector_type(4)))  int          i32x4;

#define BB   64
#define TT   600
#define UU   64
#define HH   400
#define H4   1600
#define KATT 10
#define MMIX 20
#define TXTD 60
#define NT   100            // 1600 / 16 N-tiles
#define KC_H 13             // ceil(400/32) K-chunks (padded to 416)
#define KC_W 2              // ceil(64/32)  K-chunks for window (60 -> 64)
#define HSLOT (4*KC_H*32*16) // u16 elements per packed h slot (4 m-tiles)
#define WSLOT (4*KC_W*32*16) // u16 elements per packed w slot
#define BTN  (BB*TT)
#define SMEM_B (2 * KC_H * 1024)   // bytes: 2 n-tiles x 13 chunks x 1KB

__device__ __forceinline__ unsigned short f2bf(float f) {
  union { float f; unsigned int u; } v; v.f = f;
  unsigned int u = v.u;
  u += 0x7FFFu + ((u >> 16) & 1u);
  return (unsigned short)(u >> 16);
}
__device__ __forceinline__ float sigf(float x) { return 1.0f / (1.0f + expf(-x)); }

// ---------------------------------------------------------------------------
// Pack an [N, K] row-major f32 weight matrix into WMMA B-operand bf16 tiles:
// layout [ntile][kchunk][lane(32)][16], lanes 0-15 hold K 0..7,16..23 of the chunk,
// lanes 16-31 hold K 8..15,24..31; n = ntile*16 + (lane&15). Zero-pads OOB.
// ---------------------------------------------------------------------------
__global__ void pack_b_kernel(const float* __restrict__ W, unsigned short* __restrict__ out,
                              int N, int Kdim, int ntiles, int kchunks)
{
  int idx = blockIdx.x * blockDim.x + threadIdx.x;
  int total = ntiles * kchunks * 512;
  if (idx >= total) return;
  int e    = idx & 15;
  int lane = (idx >> 4) & 31;
  int kc   = (idx >> 9) % kchunks;
  int nt   = idx / (kchunks << 9);
  int n = nt * 16 + (lane & 15);
  int k = kc * 32 + ((e < 8) ? (((lane >> 4) << 3) + e) : (16 + ((lane >> 4) << 3) + (e - 8)));
  float v = (n < N && k < Kdim) ? W[(size_t)n * Kdim + k] : 0.0f;
  out[idx] = f2bf(v);
}

// ---------------------------------------------------------------------------
// Per-step gate pre-activation GEMM: G[64x1600] = sum_seg A_seg @ B_seg^T.
// Block = 8 waves covering 4 m-tiles x 2 n-tiles, so the two B0 n-tile
// columns (26KB) are staged once per block into LDS by the Tensor Data Mover
// and reused by 4 waves each; A/window/h2 segments read straight from L2.
// Chunk counts are template constants so all WMMA loops fully unroll.
// ---------------------------------------------------------------------------
template <int NK0, int NK1, int NK2>
__global__ __launch_bounds__(256) void step_gemm_k(
    const unsigned short* __restrict__ A0, const unsigned short* __restrict__ B0,
    const unsigned short* __restrict__ A1, const unsigned short* __restrict__ B1,
    const unsigned short* __restrict__ A2, const unsigned short* __restrict__ B2,
    float* __restrict__ G)
{
  extern __shared__ __align__(64) unsigned short smemB[];
  int wave = blockIdx.x * 8 + (threadIdx.x >> 5);
  int lane = threadIdx.x & 31;
  int mtile = wave & 3;        // 4 m-tiles per n-tile pair
  int ntile = wave >> 2;       // 0..99
  int ntile0 = blockIdx.x * 2; // first n-tile staged by this block
  int lnt = ntile & 1;         // which of the two staged n-tiles is mine

  // ---- stage B0 tiles [ntile0 .. ntile0+1] into LDS ----
  const unsigned short* b0base = B0 + (size_t)ntile0 * NK0 * 512;
#if __has_builtin(__builtin_amdgcn_tensor_load_to_lds)
  if (threadIdx.x < 32) {
    unsigned long long ga = (unsigned long long)(const void*)b0base;
    const unsigned n8 = (unsigned)(2u * NK0 * 1024u) >> 3;  // 8-byte elements
    u32x4 g0;
    g0[0] = 1u;                                   // count=1 (valid user D#)
    g0[1] = 0u;                                   // lds_addr = dynamic-LDS base
    g0[2] = (unsigned)(ga & 0xFFFFFFFFu);         // global_addr[31:0]
    g0[3] = (unsigned)((ga >> 32) & 0x01FFFFFFu)  // global_addr[56:32]
          | (2u << 30);                           // type = 2 ("image")
    i32x8 g1;
    g1[0] = (int)(3u << 16);                      // data_size = 8 bytes
    g1[1] = (int)((n8 & 0xFFFFu) << 16);          // tensor_dim0[15:0]
    g1[2] = (int)(((n8 >> 16) & 0xFFFFu) | (1u << 16)); // dim0 hi | tensor_dim1=1
    g1[3] = (int)((n8 & 0xFFFFu) << 16);          // tile_dim0 = n8
    g1[4] = 1;                                    // tile_dim1 = 1
    g1[5] = (int)n8;                              // tensor_dim0_stride
    g1[6] = 0;
    g1[7] = 0;
    i32x4 z4 = {};
#if __clang_major__ >= 23
    i32x8 z8 = {};
    __builtin_amdgcn_tensor_load_to_lds(g0, g1, z4, z4, z8, 0);
#else
    __builtin_amdgcn_tensor_load_to_lds(g0, g1, z4, z4, 0);
#endif
    __builtin_amdgcn_s_wait_tensorcnt(0);
  }
  __syncthreads();
#else
  // fallback: cooperative copy through VGPRs
  {
    const int n16 = (2 * NK0 * 1024) / 16;
    for (int i = threadIdx.x; i < n16; i += 256)
      ((int4*)smemB)[i] = ((const int4*)b0base)[i];
    __syncthreads();
  }
#endif

  v8f acc = {};
#pragma unroll
  for (int kc = 0; kc < NK0; ++kc) {
    v16bf a = *(const v16bf*)(A0 + (((size_t)(mtile * NK0 + kc) * 32 + lane) << 4));
    v16bf b = *(const v16bf*)(smemB + (((size_t)(lnt * NK0 + kc) * 32 + lane) << 4));
    acc = __builtin_amdgcn_wmma_f32_16x16x32_bf16(false, a, false, b, (short)0, acc, false, false);
  }
#pragma unroll
  for (int kc = 0; kc < NK1; ++kc) {
    v16bf a = *(const v16bf*)(A1 + (((size_t)(mtile * NK1 + kc) * 32 + lane) << 4));
    v16bf b = *(const v16bf*)(B1 + (((size_t)(ntile * NK1 + kc) * 32 + lane) << 4));
    acc = __builtin_amdgcn_wmma_f32_16x16x32_bf16(false, a, false, b, (short)0, acc, false, false);
  }
  if (NK2 > 0) {
#pragma unroll
    for (int kc = 0; kc < NK2; ++kc) {
      v16bf a = *(const v16bf*)(A2 + (((size_t)(mtile * NK2 + kc) * 32 + lane) << 4));
      v16bf b = *(const v16bf*)(B2 + (((size_t)(ntile * NK2 + kc) * 32 + lane) << 4));
      acc = __builtin_amdgcn_wmma_f32_16x16x32_bf16(false, a, false, b, (short)0, acc, false, false);
    }
  }
  // C/D layout: n = lane&15, rows r -> M = r + (lane>=16 ? 8 : 0)
  int n  = ntile * 16 + (lane & 15);
  int mb = mtile * 16 + ((lane >> 4) << 3);
#pragma unroll
  for (int r = 0; r < 8; ++r)
    G[(size_t)(mb + r) * H4 + n] = acc[r];
}

// ---------------------------------------------------------------------------
// Final MDN head GEMM: pm[(t*64+b), 128] = h3[t] @ mdn_w^T + h2[t] @ Mout2^T
// waves = 600 t * 4 m-tiles * 8 n-tiles = 19200. grid 2400 x 256.
// ---------------------------------------------------------------------------
__global__ __launch_bounds__(256) void final_gemm(
    const unsigned short* __restrict__ h3p, const unsigned short* __restrict__ h2p,
    const unsigned short* __restrict__ mdnp, const unsigned short* __restrict__ mo2p,
    float* __restrict__ pm)
{
  int wave = blockIdx.x * 8 + (threadIdx.x >> 5);
  int lane = threadIdx.x & 31;
  int t = wave >> 5;
  int rem = wave & 31;
  int mt = rem >> 3;
  int ntile = rem & 7;
  const unsigned short* A3 = h3p + (size_t)(t + 1) * HSLOT;
  const unsigned short* A2 = h2p + (size_t)(t + 1) * HSLOT;
  v8f acc = {};
#pragma unroll
  for (int kc = 0; kc < KC_H; ++kc) {
    v16bf a = *(const v16bf*)(A3 + (((size_t)(mt * KC_H + kc) * 32 + lane) << 4));
    v16bf b = *(const v16bf*)(mdnp + (((size_t)(ntile * KC_H + kc) * 32 + lane) << 4));
    acc = __builtin_amdgcn_wmma_f32_16x16x32_bf16(false, a, false, b, (short)0, acc, false, false);
  }
#pragma unroll
  for (int kc = 0; kc < KC_H; ++kc) {
    v16bf a = *(const v16bf*)(A2 + (((size_t)(mt * KC_H + kc) * 32 + lane) << 4));
    v16bf b = *(const v16bf*)(mo2p + (((size_t)(ntile * KC_H + kc) * 32 + lane) << 4));
    acc = __builtin_amdgcn_wmma_f32_16x16x32_bf16(false, a, false, b, (short)0, acc, false, false);
  }
  int n  = ntile * 16 + (lane & 15);
  int mb = mt * 16 + ((lane >> 4) << 3);
#pragma unroll
  for (int r = 0; r < 8; ++r)
    pm[(size_t)(t * 64 + mb + r) * 128 + n] = acc[r];
}

// ---------------------------------------------------------------------------
// LSTM gates: g = G + folded strokes + bias (+ extra); c,h update; write h as
// packed bf16 A-operand into the next slot (ISA 16-bit A-matrix layout).
// ---------------------------------------------------------------------------
__global__ void lstm_gate(const float* __restrict__ G, float* __restrict__ c,
                          unsigned short* __restrict__ hpack, float* __restrict__ hf32,
                          const float* __restrict__ strokes_t, int strideB,
                          const float* __restrict__ Mw, const float* __restrict__ biasv,
                          const float* __restrict__ extra)
{
  int idx = blockIdx.x * blockDim.x + threadIdx.x;
  if (idx >= BB * HH) return;
  int b = idx / HH, j = idx % HH;
  float s0 = strokes_t[(size_t)b * strideB + 0];
  float s1 = strokes_t[(size_t)b * strideB + 1];
  float s2 = strokes_t[(size_t)b * strideB + 2];
  float g4[4];
#pragma unroll
  for (int q = 0; q < 4; ++q) {
    int n = j + q * HH;
    float v = G[(size_t)b * H4 + n] + biasv[n]
            + Mw[n * 3 + 0] * s0 + Mw[n * 3 + 1] * s1 + Mw[n * 3 + 2] * s2;
    if (extra) v += extra[(size_t)b * H4 + n];
    g4[q] = v;
  }
  float cn = sigf(g4[1]) * c[idx] + sigf(g4[0]) * tanhf(g4[2]);
  float h  = sigf(g4[3]) * tanhf(cn);
  c[idx] = cn;
  if (hf32) hf32[idx] = h;
  // scatter into packed A layout
  int m = b & 15, mtile = b >> 4;
  int kc = j >> 5, kk = j & 31;
  int lane = m + (((kk >> 3) & 1) << 4);
  int e = (kk & 7) + ((kk & 16) >> 1);
  hpack[(((size_t)(mtile * KC_H + kc) * 32 + lane) << 4) + e] = f2bf(h);
}

// p3k[b, 0..29] = softwin(h1)
__global__ void attn_p3k(const float* __restrict__ h1, const float* __restrict__ sww,
                         const float* __restrict__ swb, float* __restrict__ p3k)
{
  int idx = blockIdx.x * blockDim.x + threadIdx.x;
  if (idx >= BB * 30) return;
  int b = idx / 30, j = idx % 30;
  float s = swb[j];
  for (int h = 0; h < HH; ++h) s += h1[(size_t)b * HH + h] * sww[(size_t)j * HH + h];
  p3k[b * 32 + j] = s;
}

// soft window: phi over U, wt = phi @ onehots[b]; pack wt into A layout slot
__global__ void attn_win(const float* __restrict__ p3k, float* __restrict__ kp,
                         const float* __restrict__ onehots, const float* __restrict__ masks,
                         int t, unsigned short* __restrict__ wslot)
{
  __shared__ float sa[KATT], sb[KATT], sk[KATT], sphi[UU];
  int b = blockIdx.x, tid = threadIdx.x;
  if (tid < KATT) {
    sa[tid] = expf(p3k[b * 32 + tid]);
    sb[tid] = expf(p3k[b * 32 + KATT + tid]);
    sk[tid] = expf(p3k[b * 32 + 2 * KATT + tid]) + kp[b * KATT + tid];
  }
  __syncthreads();
  float mt = masks[(size_t)b * TT + t];
  float u = (float)tid;
  float phi = 0.0f;
#pragma unroll
  for (int kk = 0; kk < KATT; ++kk) {
    float d = sk[kk] - u;
    phi += sa[kk] * expf(-sb[kk] * d * d);
  }
  sphi[tid] = phi * mt;
  __syncthreads();
  if (tid < KATT) kp[b * KATT + tid] = sk[tid];
  if (tid < TXTD) {
    float w = 0.0f;
    for (int uu = 0; uu < UU; ++uu)
      w += sphi[uu] * onehots[((size_t)b * UU + uu) * TXTD + tid];
    int m = b & 15, mtile = b >> 4;
    int kc = tid >> 5, kk = tid & 31;
    int lane = m + (((kk >> 3) & 1) << 4);
    int e = (kk & 7) + ((kk & 16) >> 1);
    wslot[(((size_t)(mtile * KC_W + kc) * 32 + lane) << 4) + e] = f2bf(w);
  }
}

// pack w_prev into window slot 0
__global__ void wprev_pack(const float* __restrict__ wprev, unsigned short* __restrict__ wslot)
{
  int idx = blockIdx.x * blockDim.x + threadIdx.x;
  if (idx >= BB * TXTD) return;
  int b = idx / TXTD, d = idx % TXTD;
  int m = b & 15, mtile = b >> 4;
  int kc = d >> 5, kk = d & 31;
  int lane = m + (((kk >> 3) & 1) << 4);
  int e = (kk & 7) + ((kk & 16) >> 1);
  wslot[(((size_t)(mtile * KC_W + kc) * 32 + lane) << 4) + e] = f2bf(wprev[idx]);
}

// Wc[n, d<60] = sum_h wih[n,h] * wtr[h,d]; stride 64, zero pad d>=60
__global__ void compute_wc(const float* __restrict__ wih, const float* __restrict__ wtr,
                           float* __restrict__ out)
{
  int idx = blockIdx.x * blockDim.x + threadIdx.x;
  if (idx >= H4 * 64) return;
  int n = idx >> 6, d = idx & 63;
  float s = 0.0f;
  if (d < TXTD)
    for (int h = 0; h < HH; ++h) s += wih[(size_t)n * HH + h] * wtr[(size_t)h * TXTD + d];
  out[idx] = s;
}

// Mout2[n,h] = sum_k mdn_w[n,k] * so2_w[k,h]   ([121,400] row-major)
__global__ void compute_mout2(const float* __restrict__ mdnw, const float* __restrict__ so2w,
                              float* __restrict__ out)
{
  int idx = blockIdx.x * blockDim.x + threadIdx.x;
  if (idx >= 121 * HH) return;
  int n = idx / HH, h = idx % HH;
  float s = 0.0f;
  for (int k = 0; k < HH; ++k) s += mdnw[(size_t)n * HH + k] * so2w[(size_t)k * HH + h];
  out[idx] = s;
}

// folded bias: bias[n] = bih[n]+bhh[n] + sum_h wih[n,h]*(ba[h]+bb[h])
__global__ void bias_fold(const float* __restrict__ wih, const float* __restrict__ ba,
                          const float* __restrict__ bb, const float* __restrict__ bih,
                          const float* __restrict__ bhh, float* __restrict__ out)
{
  int n = blockIdx.x * blockDim.x + threadIdx.x;
  if (n >= H4) return;
  float s = bih[n] + bhh[n];
  for (int h = 0; h < HH; ++h) s += wih[(size_t)n * HH + h] * (ba[h] + bb[h]);
  out[n] = s;
}

// stroke fold: M[n,s] = sum_h wih[n,h]*wsm[h,s]  (wsm is [400,3])
__global__ void m_fold(const float* __restrict__ wih, const float* __restrict__ wsm,
                       float* __restrict__ out)
{
  int idx = blockIdx.x * blockDim.x + threadIdx.x;
  if (idx >= H4 * 3) return;
  int n = idx / 3, s = idx % 3;
  float acc = 0.0f;
  for (int h = 0; h < HH; ++h) acc += wih[(size_t)n * HH + h] * wsm[h * 3 + s];
  out[idx] = acc;
}

// out[b,n] = sum_h Wm[n,h] * X[b,h]   (Wm [1600,400], X [64,400])
__global__ void mv_bn(const float* __restrict__ Wm, const float* __restrict__ X,
                      float* __restrict__ out)
{
  int idx = blockIdx.x * blockDim.x + threadIdx.x;
  if (idx >= BB * H4) return;
  int b = idx / H4, n = idx % H4;
  float s = 0.0f;
  for (int h = 0; h < HH; ++h) s += Wm[(size_t)n * HH + h] * X[(size_t)b * HH + h];
  out[idx] = s;
}

// so1h[b,h] = skip_out1(h1_last)[b,h] + so1_b[h] + so2_b[h]
__global__ void so1h_kernel(const float* __restrict__ so1w, const float* __restrict__ so1b,
                            const float* __restrict__ so2b, const float* __restrict__ h1f,
                            float* __restrict__ out)
{
  int idx = blockIdx.x * blockDim.x + threadIdx.x;
  if (idx >= BB * HH) return;
  int b = idx / HH, h = idx % HH;
  float s = so1b[h] + so2b[h];
  for (int h2 = 0; h2 < HH; ++h2) s += so1w[(size_t)h * HH + h2] * h1f[(size_t)b * HH + h2];
  out[idx] = s;
}

// constv[b,n] = mdn_b[n] + sum_h mdn_w[n,h]*so1h[b,h]  (n<121, else 0)
__global__ void constv_kernel(const float* __restrict__ mdnw, const float* __restrict__ mdnb,
                              const float* __restrict__ so1h, float* __restrict__ out)
{
  int idx = blockIdx.x * blockDim.x + threadIdx.x;
  if (idx >= BB * 128) return;
  int b = idx / 128, n = idx % 128;
  float s = 0.0f;
  if (n < 121) {
    s = mdnb[n];
    for (int h = 0; h < HH; ++h) s += mdnw[(size_t)n * HH + h] * so1h[(size_t)b * HH + h];
  }
  out[idx] = s;
}

// MDN nonlinearity + output scatter into tuple-concatenated d_out
__global__ void mdn_out_kernel(const float* __restrict__ pm, const float* __restrict__ constv,
                               float* __restrict__ out)
{
  int idx = blockIdx.x * blockDim.x + threadIdx.x;
  if (idx >= BTN) return;
  int t = idx / BB, b = idx % BB;
  const float* row = pm + (size_t)idx * 128;
  const float* cv = constv + (size_t)b * 128;
  size_t o = (size_t)b * TT + t;      // output tensors are [B, T, ...]
  out[o] = sigf(row[120] + cv[120]);  // end
  float mx = -1e30f;
  for (int j = 0; j < MMIX; ++j) mx = fmaxf(mx, row[j] + cv[j]);
  float e[MMIX], s = 0.0f;
  for (int j = 0; j < MMIX; ++j) { e[j] = expf(row[j] + cv[j] - mx); s += e[j]; }
  float inv = 1.0f / s;
  float* ow  = out + BTN;
  float* om1 = ow  + (size_t)BTN * MMIX;
  float* om2 = om1 + (size_t)BTN * MMIX;
  float* os1 = om2 + (size_t)BTN * MMIX;
  float* os2 = os1 + (size_t)BTN * MMIX;
  float* opr = os2 + (size_t)BTN * MMIX;
  for (int j = 0; j < MMIX; ++j) {
    ow [o * MMIX + j] = e[j] * inv;
    om1[o * MMIX + j] = row[20 + j] + cv[20 + j];
    om2[o * MMIX + j] = row[40 + j] + cv[40 + j];
    os1[o * MMIX + j] = expf(row[60 + j] + cv[60 + j]);
    os2[o * MMIX + j] = expf(row[80 + j] + cv[80 + j]);
    opr[o * MMIX + j] = tanhf(row[100 + j] + cv[100 + j]);
  }
}

extern "C" void kernel_launch(void* const* d_in, const int* in_sizes, int n_in,
                              void* d_out, int out_size, void* d_ws, size_t ws_size,
                              hipStream_t stream)
{
  const float* strokes = (const float*)d_in[0];
  const float* onehots = (const float*)d_in[1];
  const float* masks   = (const float*)d_in[2];
  /* d_in[3] text_lens: unused by reference math */
  const float* w_prev  = (const float*)d_in[4];
  const float* in_tr_w = (const float*)d_in[5];
  const float* in_tr_b = (const float*)d_in[6];
  const float* si1_w = (const float*)d_in[7];
  const float* si1_b = (const float*)d_in[8];
  const float* si2_w = (const float*)d_in[9];
  const float* si2_b = (const float*)d_in[10];
  const float* so1_w = (const float*)d_in[11];
  const float* so1_b = (const float*)d_in[12];
  const float* so2_w = (const float*)d_in[13];
  const float* so2_b = (const float*)d_in[14];
  const float* wt1_w = (const float*)d_in[15];
  const float* wt1_b = (const float*)d_in[16];
  const float* wt2_w = (const float*)d_in[17];
  const float* wt2_b = (const float*)d_in[18];
  const float* wt3_w = (const float*)d_in[19];
  const float* wt3_b = (const float*)d_in[20];
  const float* sw_w  = (const float*)d_in[21];
  const float* sw_b  = (const float*)d_in[22];
  const float* mdn_w = (const float*)d_in[23];
  const float* mdn_b = (const float*)d_in[24];
  const float* wih1 = (const float*)d_in[25];
  const float* whh1 = (const float*)d_in[26];
  const float* bih1 = (const float*)d_in[27];
  const float* bhh1 = (const float*)d_in[28];
  const float* wih2 = (const float*)d_in[29];
  const float* whh2 = (const float*)d_in[30];
  const float* bih2 = (const float*)d_in[31];
  const float* bhh2 = (const float*)d_in[32];
  const float* wih3 = (const float*)d_in[33];
  const float* whh3 = (const float*)d_in[34];
  const float* bih3 = (const float*)d_in[35];
  const float* bhh3 = (const float*)d_in[36];

  char* p = (char*)d_ws;
  auto alloc = [&](size_t bytes) -> void* {
    void* r = (void*)p;
    p += (bytes + 255) & ~(size_t)255;
    return r;
  };
  // packed bf16 weights
  unsigned short* whh1_p = (unsigned short*)alloc((size_t)NT * KC_H * 512 * 2);
  unsigned short* whh2_p = (unsigned short*)alloc((size_t)NT * KC_H * 512 * 2);
  unsigned short* whh3_p = (unsigned short*)alloc((size_t)NT * KC_H * 512 * 2);
  unsigned short* wih3_p = (unsigned short*)alloc((size_t)NT * KC_H * 512 * 2);
  unsigned short* wc1_p  = (unsigned short*)alloc((size_t)NT * KC_W * 512 * 2);
  unsigned short* wc2_p  = (unsigned short*)alloc((size_t)NT * KC_W * 512 * 2);
  unsigned short* wc3_p  = (unsigned short*)alloc((size_t)NT * KC_W * 512 * 2);
  unsigned short* mdn_p  = (unsigned short*)alloc((size_t)8 * KC_H * 512 * 2);
  unsigned short* mo2_p  = (unsigned short*)alloc((size_t)8 * KC_H * 512 * 2);
  // packed activations
  unsigned short* hp1 = (unsigned short*)alloc((size_t)2 * HSLOT * 2);           // ring
  unsigned short* hp2 = (unsigned short*)alloc((size_t)(TT + 1) * HSLOT * 2);
  unsigned short* hp3 = (unsigned short*)alloc((size_t)(TT + 1) * HSLOT * 2);
  unsigned short* wp  = (unsigned short*)alloc((size_t)(TT + 1) * WSLOT * 2);
  // f32 scratch
  float* G     = (float*)alloc((size_t)BB * H4 * 4);
  float* c1    = (float*)alloc((size_t)BB * HH * 4);
  float* c2    = (float*)alloc((size_t)BB * HH * 4);
  float* c3    = (float*)alloc((size_t)BB * HH * 4);
  float* h1f   = (float*)alloc((size_t)BB * HH * 4);
  float* kp    = (float*)alloc((size_t)BB * KATT * 4);
  float* p3k   = (float*)alloc((size_t)BB * 32 * 4);
  float* M1f   = (float*)alloc((size_t)H4 * 3 * 4);
  float* M2f   = (float*)alloc((size_t)H4 * 3 * 4);
  float* M3f   = (float*)alloc((size_t)H4 * 3 * 4);
  float* bias1 = (float*)alloc((size_t)H4 * 4);
  float* bias2 = (float*)alloc((size_t)H4 * 4);
  float* bias3 = (float*)alloc((size_t)H4 * 4);
  float* c2base= (float*)alloc((size_t)BB * H4 * 4);
  float* so1h  = (float*)alloc((size_t)BB * HH * 4);
  float* constv= (float*)alloc((size_t)BB * 128 * 4);
  float* tmp   = (float*)alloc((size_t)H4 * 64 * 4);   // reused: Wc f32 / Mout2 f32
  float* pm    = (float*)alloc((size_t)BTN * 128 * 4);

  // ---- init (every call; deterministic) ----
  hipMemsetAsync(hp1, 0, (size_t)2 * HSLOT * 2, stream);
  hipMemsetAsync(hp2, 0, (size_t)(TT + 1) * HSLOT * 2, stream);
  hipMemsetAsync(hp3, 0, (size_t)(TT + 1) * HSLOT * 2, stream);
  hipMemsetAsync(wp,  0, (size_t)(TT + 1) * WSLOT * 2, stream);
  hipMemsetAsync(c1, 0, (size_t)BB * HH * 4, stream);
  hipMemsetAsync(c2, 0, (size_t)BB * HH * 4, stream);
  hipMemsetAsync(c3, 0, (size_t)BB * HH * 4, stream);
  hipMemsetAsync(kp, 0, (size_t)BB * KATT * 4, stream);

  // ---- weight packing / folds ----
  int pbThreads = NT * KC_H * 512;
  int pbBlocks = (pbThreads + 255) / 256;
  pack_b_kernel<<<pbBlocks, 256, 0, stream>>>(whh1, whh1_p, H4, HH, NT, KC_H);
  pack_b_kernel<<<pbBlocks, 256, 0, stream>>>(whh2, whh2_p, H4, HH, NT, KC_H);
  pack_b_kernel<<<pbBlocks, 256, 0, stream>>>(whh3, whh3_p, H4, HH, NT, KC_H);
  pack_b_kernel<<<pbBlocks, 256, 0, stream>>>(wih3, wih3_p, H4, HH, NT, KC_H);

  int wcBlocks = (H4 * 64 + 255) / 256;
  int pwThreads = NT * KC_W * 512;
  int pwBlocks = (pwThreads + 255) / 256;
  compute_wc<<<wcBlocks, 256, 0, stream>>>(wih1, wt1_w, tmp);
  pack_b_kernel<<<pwBlocks, 256, 0, stream>>>(tmp, wc1_p, H4, 64, NT, KC_W);
  compute_wc<<<wcBlocks, 256, 0, stream>>>(wih2, wt2_w, tmp);
  pack_b_kernel<<<pwBlocks, 256, 0, stream>>>(tmp, wc2_p, H4, 64, NT, KC_W);
  compute_wc<<<wcBlocks, 256, 0, stream>>>(wih3, wt3_w, tmp);
  pack_b_kernel<<<pwBlocks, 256, 0, stream>>>(tmp, wc3_p, H4, 64, NT, KC_W);

  int poThreads = 8 * KC_H * 512;
  int poBlocks = (poThreads + 255) / 256;
  compute_mout2<<<(121 * HH + 255) / 256, 256, 0, stream>>>(mdn_w, so2_w, tmp);
  pack_b_kernel<<<poBlocks, 256, 0, stream>>>(tmp, mo2_p, 121, HH, 8, KC_H);
  pack_b_kernel<<<poBlocks, 256, 0, stream>>>(mdn_w, mdn_p, 121, HH, 8, KC_H);

  bias_fold<<<(H4 + 255) / 256, 256, 0, stream>>>(wih1, in_tr_b, wt1_b, bih1, bhh1, bias1);
  bias_fold<<<(H4 + 255) / 256, 256, 0, stream>>>(wih2, si1_b, wt2_b, bih2, bhh2, bias2);
  bias_fold<<<(H4 + 255) / 256, 256, 0, stream>>>(wih3, si2_b, wt3_b, bih3, bhh3, bias3);
  m_fold<<<(H4 * 3 + 255) / 256, 256, 0, stream>>>(wih1, in_tr_w, M1f);
  m_fold<<<(H4 * 3 + 255) / 256, 256, 0, stream>>>(wih2, si1_w, M2f);
  m_fold<<<(H4 * 3 + 255) / 256, 256, 0, stream>>>(wih3, si2_w, M3f);
  wprev_pack<<<(BB * TXTD + 255) / 256, 256, 0, stream>>>(w_prev, wp);

  // ---- phase 1: attention LSTM scan ----
  for (int t = 0; t < TT; ++t) {
    step_gemm_k<KC_H, KC_W, 0><<<50, 256, SMEM_B, stream>>>(
        hp1 + (size_t)(t & 1) * HSLOT, whh1_p,
        wp + (size_t)t * WSLOT, wc1_p,
        nullptr, nullptr, G);
    lstm_gate<<<100, 256, 0, stream>>>(
        G, c1, hp1 + (size_t)((t + 1) & 1) * HSLOT, h1f,
        strokes + (size_t)t * 3, TT * 3, M1f, bias1, nullptr);
    attn_p3k<<<8, 256, 0, stream>>>(h1f, sw_w, sw_b, p3k);
    attn_win<<<64, 64, 0, stream>>>(p3k, kp, onehots, masks, t,
                                    wp + (size_t)(t + 1) * WSLOT);
  }

  // ---- h1_last-derived constants ----
  mv_bn<<<(BB * H4 + 255) / 256, 256, 0, stream>>>(wih2, h1f, c2base);
  so1h_kernel<<<(BB * HH + 255) / 256, 256, 0, stream>>>(so1_w, so1_b, so2_b, h1f, so1h);
  constv_kernel<<<(BB * 128 + 255) / 256, 256, 0, stream>>>(mdn_w, mdn_b, so1h, constv);

  // ---- phase 2: LSTM2 scan ----
  for (int t = 0; t < TT; ++t) {
    step_gemm_k<KC_H, KC_W, 0><<<50, 256, SMEM_B, stream>>>(
        hp2 + (size_t)t * HSLOT, whh2_p,
        wp + (size_t)(t + 1) * WSLOT, wc2_p,
        nullptr, nullptr, G);
    lstm_gate<<<100, 256, 0, stream>>>(
        G, c2, hp2 + (size_t)(t + 1) * HSLOT, nullptr,
        strokes + (size_t)t * 3, TT * 3, M2f, bias2, c2base);
  }

  // ---- phase 3: LSTM3 scan (extra WMMA segment: h2[t] @ wih3^T) ----
  for (int t = 0; t < TT; ++t) {
    step_gemm_k<KC_H, KC_W, KC_H><<<50, 256, SMEM_B, stream>>>(
        hp3 + (size_t)t * HSLOT, whh3_p,
        wp + (size_t)(t + 1) * WSLOT, wc3_p,
        hp2 + (size_t)(t + 1) * HSLOT, wih3_p, G);
    lstm_gate<<<100, 256, 0, stream>>>(
        G, c3, hp3 + (size_t)(t + 1) * HSLOT, nullptr,
        strokes + (size_t)t * 3, TT * 3, M3f, bias3, nullptr);
  }

  // ---- MDN head ----
  final_gemm<<<2400, 256, 0, stream>>>(hp3, hp2, mdn_p, mo2_p, pm);
  mdn_out_kernel<<<(BTN + 255) / 256, 256, 0, stream>>>(pm, constv, (float*)d_out);
}
